// ComplexMultiheadAttention_90228672955128
// MI455X (gfx1250) — matrix operand order repaired and checked by reading, not compiled, not run on hardware
//
#include <hip/hip_runtime.h>
#include <hip/hip_bf16.h>
#include <stdint.h>

#define S_LEN   2048
#define D_MODEL 1024
#define NHEADS  16
#define DHEAD   64
#define NEG_BIG (-3.0e38f)
#define KLD     72   // K-tile LDS row stride (ushorts): 16B multiple for async DMA

typedef __attribute__((ext_vector_type(16))) __bf16 bf16x16;
typedef __attribute__((ext_vector_type(8)))  float  f32x8;

#if defined(__gfx1250__) && __has_builtin(__builtin_amdgcn_global_load_async_to_lds_b128)
#define HAVE_ASYNC_LDS 1
#else
#define HAVE_ASYNC_LDS 0
#endif

union Frag {
  bf16x16  v;
  uint32_t d[8];
  unsigned short u[16];
};

__device__ __forceinline__ f32x8 vzero() {
  f32x8 z;
#pragma unroll
  for (int i = 0; i < 8; ++i) z[i] = 0.0f;
  return z;
}

#if __has_builtin(__builtin_amdgcn_cvt_pk_bf16_f32)
__device__ __forceinline__ uint32_t f2bf2(float lo, float hi) {
  typedef __attribute__((ext_vector_type(2))) __bf16 bf16x2;
  union { bf16x2 v; uint32_t u; } c;
  c.v = __builtin_amdgcn_cvt_pk_bf16_f32(lo, hi);
  return c.u;
}
__device__ __forceinline__ unsigned short f2bf(float f) {
  return (unsigned short)(f2bf2(f, 0.0f) & 0xFFFFu);
}
#else
__device__ __forceinline__ unsigned short f2bf(float f) {
  union { float f; uint32_t u; } c; c.f = f;
  uint32_t u = c.u + 0x7FFFu + ((c.u >> 16) & 1u);   // round-to-nearest-even
  return (unsigned short)(u >> 16);
}
__device__ __forceinline__ uint32_t f2bf2(float lo, float hi) {
  return (uint32_t)f2bf(lo) | ((uint32_t)f2bf(hi) << 16);
}
#endif

#if HAVE_ASYNC_LDS
// Builtin signature (from hipcc diagnostic): param 1 is global pointer to
// 'int __vector(4)'; pass both pointers as int4-vector pointers in their
// address spaces via uintptr_t (generic->as(3) keeps the low-32 LDS offset).
typedef __attribute__((ext_vector_type(4))) int int4v;
typedef __attribute__((address_space(1))) int4v* g_int4p;
typedef __attribute__((address_space(3))) int4v* l_int4p;
__device__ __forceinline__ void async_cp16(unsigned short* l, const unsigned short* g) {
  __builtin_amdgcn_global_load_async_to_lds_b128(
      (g_int4p)(uintptr_t)g, (l_int4p)(uintptr_t)l, 0, 0);
}
#endif

// A fragment (M16 x K32 bf16). rowptr = &tile[this lane's M row][k0].
// Layout: lanes 0-15 hold K=0..15, lanes 16-31 hold K=16..31 (matches frag_b_ptr).
__device__ __forceinline__ Frag frag_a_ptr(const unsigned short* rowptr, int lane) {
  Frag f;
  const unsigned short* p = rowptr + ((lane & 16) ? 16 : 0);
#pragma unroll
  for (int v = 0; v < 4; ++v) f.d[v]     = *(const uint32_t*)(p + 2 * v);
#pragma unroll
  for (int v = 0; v < 4; ++v) f.d[4 + v] = *(const uint32_t*)(p + 8 + 2 * v);
  return f;
}

// B fragment (K32 x N16 bf16) from an [N][K]-stored tile. rowptr = &tile[this lane's N][k0].
// Layout (contiguous k, same as frag_a_ptr): lanes 0-15 hold K=0..15 (N=lane), lanes 16-31 hold K=16..31 (N=lane-16).
__device__ __forceinline__ Frag frag_b_ptr(const unsigned short* rowptr, int lane) {
  Frag f;
  const unsigned short* p = rowptr + ((lane & 16) ? 16 : 0);
#pragma unroll
  for (int v = 0; v < 8; ++v) f.d[v] = *(const uint32_t*)(p + 2 * v);
  return f;
}

__device__ __forceinline__ Frag fneg(Frag f) {
#pragma unroll
  for (int v = 0; v < 8; ++v) f.d[v] ^= 0x80008000u;
  return f;
}

__device__ __forceinline__ f32x8 wmma_bf16(const Frag& a, const Frag& b, f32x8 c) {
  return __builtin_amdgcn_wmma_f32_16x16x32_bf16(false, a.v, false, b.v,
                                                 (short)0, c, false, false);
}

// ---------------------------------------------------------------------------
// Kernel 1: real GEMM  dst(bf16,[H][S][Dh]) = X(f32,[S][D]) @ W(f32,[D][D])^T
// 128x128 block tile, BK=32, 8 waves each 32x64.
// ---------------------------------------------------------------------------
__global__ __launch_bounds__(256) void k_proj(const float* __restrict__ X,
                                              const float* __restrict__ W,
                                              unsigned short* __restrict__ dst) {
  __shared__ __align__(16) unsigned short sA[128 * 34];
  __shared__ __align__(16) unsigned short sB[128 * 34];

  const int t    = threadIdx.x;
  const int lane = t & 31, wid = t >> 5;
  const int wm   = (wid & 3) * 32;   // wave M offset (4 waves along M)
  const int wn   = (wid >> 2) * 64;  // wave N offset (2 waves along N)
  const int m0   = blockIdx.x * 128, n0 = blockIdx.y * 128;
  const int row  = t >> 1, c0 = (t & 1) * 16;

  f32x8 acc[2][4];
#pragma unroll
  for (int mi = 0; mi < 2; ++mi)
#pragma unroll
    for (int ni = 0; ni < 4; ++ni) acc[mi][ni] = vzero();

  for (int k0 = 0; k0 < D_MODEL; k0 += 32) {
    __syncthreads();
    {
      const float4* ga = (const float4*)(X + (size_t)(m0 + row) * D_MODEL + k0 + c0);
      float4 a0 = ga[0], a1 = ga[1], a2 = ga[2], a3 = ga[3];
      uint32_t* da = (uint32_t*)&sA[row * 34 + c0];
      da[0] = f2bf2(a0.x, a0.y); da[1] = f2bf2(a0.z, a0.w);
      da[2] = f2bf2(a1.x, a1.y); da[3] = f2bf2(a1.z, a1.w);
      da[4] = f2bf2(a2.x, a2.y); da[5] = f2bf2(a2.z, a2.w);
      da[6] = f2bf2(a3.x, a3.y); da[7] = f2bf2(a3.z, a3.w);
      const float4* gb = (const float4*)(W + (size_t)(n0 + row) * D_MODEL + k0 + c0);
      float4 b0 = gb[0], b1 = gb[1], b2 = gb[2], b3 = gb[3];
      uint32_t* db = (uint32_t*)&sB[row * 34 + c0];
      db[0] = f2bf2(b0.x, b0.y); db[1] = f2bf2(b0.z, b0.w);
      db[2] = f2bf2(b1.x, b1.y); db[3] = f2bf2(b1.z, b1.w);
      db[4] = f2bf2(b2.x, b2.y); db[5] = f2bf2(b2.z, b2.w);
      db[6] = f2bf2(b3.x, b3.y); db[7] = f2bf2(b3.z, b3.w);
      if (k0 + 32 < D_MODEL) {
        __builtin_prefetch(X + (size_t)(m0 + row) * D_MODEL + k0 + 32 + c0, 0, 1);
        __builtin_prefetch(W + (size_t)(n0 + row) * D_MODEL + k0 + 32 + c0, 0, 1);
      }
    }
    __syncthreads();

    Frag fa[2], fb[4];
#pragma unroll
    for (int mi = 0; mi < 2; ++mi)
      fa[mi] = frag_a_ptr(&sA[(wm + mi * 16 + (lane & 15)) * 34], lane);
#pragma unroll
    for (int ni = 0; ni < 4; ++ni)
      fb[ni] = frag_b_ptr(&sB[(wn + ni * 16 + (lane & 15)) * 34], lane);
#pragma unroll
    for (int mi = 0; mi < 2; ++mi)
#pragma unroll
      for (int ni = 0; ni < 4; ++ni)
        acc[mi][ni] = wmma_bf16(fa[mi], fb[ni], acc[mi][ni]);
  }

  const int rbase = (lane & 16) ? 8 : 0;
#pragma unroll
  for (int mi = 0; mi < 2; ++mi)
#pragma unroll
    for (int ni = 0; ni < 4; ++ni)
#pragma unroll
      for (int r = 0; r < 8; ++r) {
        int s = m0 + wm + mi * 16 + rbase + r;
        int j = n0 + wn + ni * 16 + (lane & 15);
        int h = j >> 6, dh = j & 63;
        dst[(size_t)(h * S_LEN + s) * DHEAD + dh] = f2bf(acc[mi][ni][r]);
      }
}

// ---------------------------------------------------------------------------
// Kernel 2: complex flash attention per (head, 64 query rows).
// 4 waves, each owns 16 query rows. 32-key steps, online magnitude softmax,
// phase rotation, complex P@V accumulated in registers via WMMA.
// K tiles staged via async DMA to LDS when available (ASYNCcnt path).
// ---------------------------------------------------------------------------
__global__ __launch_bounds__(128) void k_attn(const unsigned short* __restrict__ qkv,
                                              unsigned short* __restrict__ attR,
                                              unsigned short* __restrict__ attI) {
  __shared__ __align__(16) unsigned short sKr[32 * KLD];
  __shared__ __align__(16) unsigned short sKi[32 * KLD];
  __shared__ __align__(16) unsigned short sVr[64 * 34];   // transposed [dh][key]
  __shared__ __align__(16) unsigned short sVi[64 * 34];
  __shared__ __align__(16) unsigned short sPr[4][16 * 34];
  __shared__ __align__(16) unsigned short sPi[4][16 * 34];

  const int t    = threadIdx.x;
  const int lane = t & 31, wid = t >> 5;
  const int h    = blockIdx.y;
  const int q0   = blockIdx.x * 64;
  const int q0w  = q0 + wid * 16;
  const int col  = lane & 15;
  const int rbase = (lane & 16) ? 8 : 0;

  const size_t SEG = (size_t)NHEADS * S_LEN * DHEAD;
  const unsigned short* Qr = qkv + 0 * SEG;
  const unsigned short* Qi = qkv + 1 * SEG;
  const unsigned short* Kr = qkv + 2 * SEG;
  const unsigned short* Ki = qkv + 3 * SEG;
  const unsigned short* Vr = qkv + 4 * SEG;
  const unsigned short* Vi = qkv + 5 * SEG;

  // Loop-invariant Q fragments (A layout, Dh split as K=0..31 and K=32..63)
  const unsigned short* qrow_r = Qr + (size_t)(h * S_LEN + q0w + col) * DHEAD;
  const unsigned short* qrow_i = Qi + (size_t)(h * S_LEN + q0w + col) * DHEAD;
  Frag fqr[2], fqi[2];
  fqr[0] = frag_a_ptr(qrow_r, lane);      fqr[1] = frag_a_ptr(qrow_r + 32, lane);
  fqi[0] = frag_a_ptr(qrow_i, lane);      fqi[1] = frag_a_ptr(qrow_i + 32, lane);

  f32x8 Or[4], Oi[4];
#pragma unroll
  for (int nt = 0; nt < 4; ++nt) { Or[nt] = vzero(); Oi[nt] = vzero(); }
  float mrun[8], lrun[8];
#pragma unroll
  for (int r = 0; r < 8; ++r) { mrun[r] = NEG_BIG; lrun[r] = 0.0f; }

  const int srow = t >> 2;         // 0..31 (key index for staging)
  const int sc0  = (t & 3) * 16;   // 0/16/32/48 (dh chunk for staging)
  const int kend = q0 + 64;        // causal: no keys beyond last query of block

  for (int kb = 0; kb < kend; kb += 32) {
    __syncthreads();
    {
      const size_t gbase = (size_t)(h * S_LEN + kb + srow) * DHEAD + sc0;
#if HAVE_ASYNC_LDS
      // DMA K tiles straight into LDS (tracked by ASYNCcnt, bypasses VGPRs)
      async_cp16(&sKr[srow * KLD + sc0],     Kr + gbase);
      async_cp16(&sKr[srow * KLD + sc0 + 8], Kr + gbase + 8);
      async_cp16(&sKi[srow * KLD + sc0],     Ki + gbase);
      async_cp16(&sKi[srow * KLD + sc0 + 8], Ki + gbase + 8);
#else
      const uint32_t* gkr = (const uint32_t*)(Kr + gbase);
      const uint32_t* gki = (const uint32_t*)(Ki + gbase);
      uint32_t* dkr = (uint32_t*)&sKr[srow * KLD + sc0];
      uint32_t* dki = (uint32_t*)&sKi[srow * KLD + sc0];
#pragma unroll
      for (int j = 0; j < 8; ++j) { dkr[j] = gkr[j]; dki[j] = gki[j]; }
#endif
      // V tiles need a transpose -> manual staging
      const unsigned short* gvr = Vr + gbase;
      const unsigned short* gvi = Vi + gbase;
#pragma unroll
      for (int j = 0; j < 16; ++j) {
        sVr[(sc0 + j) * 34 + srow] = gvr[j];
        sVi[(sc0 + j) * 34 + srow] = gvi[j];
      }
#if HAVE_ASYNC_LDS
      asm volatile("s_wait_asynccnt 0" ::: "memory");
#endif
    }
    __syncthreads();

    // ---- Hermitian scores for two 16-key tiles -------------------------
    f32x8 hr[2], hi[2];
#pragma unroll
    for (int tt = 0; tt < 2; ++tt) {
      const unsigned short* kr_row = &sKr[(tt * 16 + col) * KLD];
      const unsigned short* ki_row = &sKi[(tt * 16 + col) * KLD];
      Frag br0 = frag_b_ptr(kr_row, lane);
      Frag br1 = frag_b_ptr(kr_row + 32, lane);
      Frag bi0 = frag_b_ptr(ki_row, lane);
      Frag bi1 = frag_b_ptr(ki_row + 32, lane);
      f32x8 ar = wmma_bf16(fqr[0], br0, vzero());
      ar = wmma_bf16(fqr[1], br1, ar);
      ar = wmma_bf16(fqi[0], bi0, ar);
      ar = wmma_bf16(fqi[1], bi1, ar);           // Re = QrKr + QiKi
      hr[tt] = ar;
      Frag bi0n = fneg(bi0), bi1n = fneg(bi1);
      f32x8 ai = wmma_bf16(fqi[0], br0, vzero());
      ai = wmma_bf16(fqi[1], br1, ai);
      ai = wmma_bf16(fqr[0], bi0n, ai);
      ai = wmma_bf16(fqr[1], bi1n, ai);          // Im = QiKr - QrKi
      hi[tt] = ai;
    }

    // ---- magnitude + causal mask + online softmax ----------------------
    float ab[2][8], mg[2][8];
#pragma unroll
    for (int tt = 0; tt < 2; ++tt)
#pragma unroll
      for (int r = 0; r < 8; ++r) {
        float re = hr[tt][r], im = hi[tt][r];
        float a = sqrtf(re * re + im * im);
        ab[tt][r] = a;
        int kidx = kb + tt * 16 + col;
        int qidx = q0w + rbase + r;
        mg[tt][r] = (kidx <= qidx) ? a * 0.125f : NEG_BIG;   // 1/sqrt(64)
      }
    float corr[8];
#pragma unroll
    for (int r = 0; r < 8; ++r) {
      float m = fmaxf(mg[0][r], mg[1][r]);
      m = fmaxf(m, __shfl_xor(m, 1, 32));
      m = fmaxf(m, __shfl_xor(m, 2, 32));
      m = fmaxf(m, __shfl_xor(m, 4, 32));
      m = fmaxf(m, __shfl_xor(m, 8, 32));
      float mnew = fmaxf(mrun[r], m);
      corr[r] = __expf(mrun[r] - mnew);
      mrun[r] = mnew;
    }
    float p[2][8];
#pragma unroll
    for (int tt = 0; tt < 2; ++tt)
#pragma unroll
      for (int r = 0; r < 8; ++r) p[tt][r] = __expf(mg[tt][r] - mrun[r]);
#pragma unroll
    for (int r = 0; r < 8; ++r) {
      float s = p[0][r] + p[1][r];
      s += __shfl_xor(s, 1, 32);
      s += __shfl_xor(s, 2, 32);
      s += __shfl_xor(s, 4, 32);
      s += __shfl_xor(s, 8, 32);
      lrun[r] = lrun[r] * corr[r] + s;
    }
#pragma unroll
    for (int nt = 0; nt < 4; ++nt)
#pragma unroll
      for (int r = 0; r < 8; ++r) { Or[nt][r] *= corr[r]; Oi[nt][r] *= corr[r]; }

    // ---- phase-rotated probabilities -> per-wave LDS (C-layout scatter) -
    unsigned short* pr = sPr[wid];
    unsigned short* pi = sPi[wid];
#pragma unroll
    for (int tt = 0; tt < 2; ++tt)
#pragma unroll
      for (int r = 0; r < 8; ++r) {
        float inv = p[tt][r] / fmaxf(ab[tt][r], 1e-8f);
        int idx = (rbase + r) * 34 + tt * 16 + col;
        pr[idx] = f2bf(hr[tt][r] * inv);
        pi[idx] = f2bf(hi[tt][r] * inv);
      }
    asm volatile("s_wait_dscnt 0" ::: "memory");  // same-wave LDS turnaround

    // ---- complex P @ V --------------------------------------------------
    Frag fpr  = frag_a_ptr(&pr[col * 34], lane);
    Frag fpi  = frag_a_ptr(&pi[col * 34], lane);
    Frag fpin = fneg(fpi);
#pragma unroll
    for (int nt = 0; nt < 4; ++nt) {
      Frag bvr = frag_b_ptr(&sVr[(nt * 16 + col) * 34], lane);
      Frag bvi = frag_b_ptr(&sVi[(nt * 16 + col) * 34], lane);
      Or[nt] = wmma_bf16(fpr,  bvr, Or[nt]);
      Or[nt] = wmma_bf16(fpin, bvi, Or[nt]);     // Re = PrVr - PiVi
      Oi[nt] = wmma_bf16(fpr,  bvi, Oi[nt]);
      Oi[nt] = wmma_bf16(fpi,  bvr, Oi[nt]);     // Im = PrVi + PiVr
    }
  }

  // ---- normalize + store to [S][D] bf16 --------------------------------
#pragma unroll
  for (int r = 0; r < 8; ++r) {
    float invl = 1.0f / lrun[r];
    int s = q0w + rbase + r;
#pragma unroll
    for (int nt = 0; nt < 4; ++nt) {
      int j = h * DHEAD + nt * 16 + col;
      attR[(size_t)s * D_MODEL + j] = f2bf(Or[nt][r] * invl);
      attI[(size_t)s * D_MODEL + j] = f2bf(Oi[nt][r] * invl);
    }
  }
}

// ---------------------------------------------------------------------------
// Kernel 3: one half of the complex output projection (f32 out, interleaved).
//   C = Ar @ B1^T + (negB2 ? -1 : +1) * Ai @ B2^T ;  out[(s*D + j)*2 + zi] = C
// ---------------------------------------------------------------------------
__global__ __launch_bounds__(256) void k_oproj(const unsigned short* __restrict__ Ar,
                                               const unsigned short* __restrict__ Ai,
                                               const float* __restrict__ B1,
                                               const float* __restrict__ B2,
                                               float* __restrict__ out,
                                               int negB2, int zi) {
  __shared__ __align__(16) unsigned short sAr[128 * 34];
  __shared__ __align__(16) unsigned short sAi[128 * 34];
  __shared__ __align__(16) unsigned short sB1[128 * 34];
  __shared__ __align__(16) unsigned short sB2[128 * 34];

  const int t    = threadIdx.x;
  const int lane = t & 31, wid = t >> 5;
  const int wm   = (wid & 3) * 32, wn = (wid >> 2) * 64;
  const int m0   = blockIdx.x * 128, n0 = blockIdx.y * 128;
  const int row  = t >> 1, c0 = (t & 1) * 16;

  f32x8 acc[2][4];
#pragma unroll
  for (int mi = 0; mi < 2; ++mi)
#pragma unroll
    for (int ni = 0; ni < 4; ++ni) acc[mi][ni] = vzero();

  for (int k0 = 0; k0 < D_MODEL; k0 += 32) {
    __syncthreads();
    {
      const uint32_t* gar = (const uint32_t*)(Ar + (size_t)(m0 + row) * D_MODEL + k0 + c0);
      const uint32_t* gai = (const uint32_t*)(Ai + (size_t)(m0 + row) * D_MODEL + k0 + c0);
      uint32_t* dar = (uint32_t*)&sAr[row * 34 + c0];
      uint32_t* dai = (uint32_t*)&sAi[row * 34 + c0];
#pragma unroll
      for (int j = 0; j < 8; ++j) { dar[j] = gar[j]; dai[j] = gai[j]; }
      const float4* g1 = (const float4*)(B1 + (size_t)(n0 + row) * D_MODEL + k0 + c0);
      const float4* g2 = (const float4*)(B2 + (size_t)(n0 + row) * D_MODEL + k0 + c0);
      float4 b0 = g1[0], b1 = g1[1], b2 = g1[2], b3 = g1[3];
      uint32_t* d1 = (uint32_t*)&sB1[row * 34 + c0];
      d1[0] = f2bf2(b0.x, b0.y); d1[1] = f2bf2(b0.z, b0.w);
      d1[2] = f2bf2(b1.x, b1.y); d1[3] = f2bf2(b1.z, b1.w);
      d1[4] = f2bf2(b2.x, b2.y); d1[5] = f2bf2(b2.z, b2.w);
      d1[6] = f2bf2(b3.x, b3.y); d1[7] = f2bf2(b3.z, b3.w);
      float4 c4_0 = g2[0], c4_1 = g2[1], c4_2 = g2[2], c4_3 = g2[3];
      uint32_t* d2 = (uint32_t*)&sB2[row * 34 + c0];
      d2[0] = f2bf2(c4_0.x, c4_0.y); d2[1] = f2bf2(c4_0.z, c4_0.w);
      d2[2] = f2bf2(c4_1.x, c4_1.y); d2[3] = f2bf2(c4_1.z, c4_1.w);
      d2[4] = f2bf2(c4_2.x, c4_2.y); d2[5] = f2bf2(c4_2.z, c4_2.w);
      d2[6] = f2bf2(c4_3.x, c4_3.y); d2[7] = f2bf2(c4_3.z, c4_3.w);
    }
    __syncthreads();

    Frag far[2], fai[2], fb1[4], fb2[4];
#pragma unroll
    for (int mi = 0; mi < 2; ++mi) {
      far[mi] = frag_a_ptr(&sAr[(wm + mi * 16 + (lane & 15)) * 34], lane);
      Frag fi = frag_a_ptr(&sAi[(wm + mi * 16 + (lane & 15)) * 34], lane);
      fai[mi] = negB2 ? fneg(fi) : fi;
    }
#pragma unroll
    for (int ni = 0; ni < 4; ++ni) {
      fb1[ni] = frag_b_ptr(&sB1[(wn + ni * 16 + (lane & 15)) * 34], lane);
      fb2[ni] = frag_b_ptr(&sB2[(wn + ni * 16 + (lane & 15)) * 34], lane);
    }
#pragma unroll
    for (int mi = 0; mi < 2; ++mi)
#pragma unroll
      for (int ni = 0; ni < 4; ++ni) {
        acc[mi][ni] = wmma_bf16(far[mi], fb1[ni], acc[mi][ni]);
        acc[mi][ni] = wmma_bf16(fai[mi], fb2[ni], acc[mi][ni]);
      }
  }

  const int rbase = (lane & 16) ? 8 : 0;
#pragma unroll
  for (int mi = 0; mi < 2; ++mi)
#pragma unroll
    for (int ni = 0; ni < 4; ++ni)
#pragma unroll
      for (int r = 0; r < 8; ++r) {
        int s = m0 + wm + mi * 16 + rbase + r;
        int j = n0 + wn + ni * 16 + (lane & 15);
        out[((size_t)s * D_MODEL + j) * 2 + zi] = acc[mi][ni][r];
      }
}

// ---------------------------------------------------------------------------
extern "C" void kernel_launch(void* const* d_in, const int* in_sizes, int n_in,
                              void* d_out, int out_size, void* d_ws, size_t ws_size,
                              hipStream_t stream) {
  const float* hidden = (const float*)d_in[0];
  // d_in order: hidden, wq_r, wq_i, wk_r, wk_i, wv_r, wv_i, wo_r, wo_i
  unsigned short* ws  = (unsigned short*)d_ws;
  const size_t SEG = (size_t)NHEADS * S_LEN * DHEAD;   // 2,097,152 elems
  unsigned short* qkv = ws;                            // 6 segments: Qr Qi Kr Ki Vr Vi
  unsigned short* aR  = ws + 6 * SEG;                  // attn real (bf16, [S][D])
  unsigned short* aI  = ws + 7 * SEG;                  // attn imag

  dim3 blk(256);
  dim3 grid(S_LEN / 128, D_MODEL / 128);
  for (int z = 0; z < 6; ++z)
    k_proj<<<grid, blk, 0, stream>>>(hidden, (const float*)d_in[1 + z],
                                     qkv + (size_t)z * SEG);

  k_attn<<<dim3(S_LEN / 64, NHEADS), 128, 0, stream>>>(qkv, aR, aI);

  const float* wo_r = (const float*)d_in[7];
  const float* wo_i = (const float*)d_in[8];
  float* out = (float*)d_out;
  // real: Ar*Wr - Ai*Wi ; imag: Ar*Wi + Ai*Wr
  k_oproj<<<grid, blk, 0, stream>>>(aR, aI, wo_r, wo_i, out, 1, 0);
  k_oproj<<<grid, blk, 0, stream>>>(aR, aI, wo_i, wo_r, out, 0, 1);
}
